// ParticleNet_6356551598518
// MI455X (gfx1250) — matrix operand order repaired and checked by reading, not compiled, not run on hardware
//
#include <hip/hip_runtime.h>

typedef __attribute__((ext_vector_type(16))) _Float16 v16h;
typedef __attribute__((ext_vector_type(8)))  _Float16 v8h;
typedef __attribute__((ext_vector_type(8)))  float    v8f;
typedef int v4i __attribute__((vector_size(16)));

#define GEMM_MBLK 128
#define GEMM_NBLK 64
#define GEMM_KSTEP 32
#define LDSW 40   // padded row stride in halves: 80B = 5*16B -> keeps 16B alignment

// ---- async Global->LDS staging (gfx1250 ASYNCcnt path), with safe fallback ----
#if defined(__has_builtin)
#if __has_builtin(__builtin_amdgcn_global_load_async_to_lds_b128)
#define ASYNC_LDS 1
#endif
#endif
#ifndef ASYNC_LDS
#define ASYNC_LDS 0
#endif

__device__ __forceinline__ void stage16B(const _Float16* __restrict__ g, _Float16* l)
{
#if ASYNC_LDS
    // builtin signature: (v4i AS1* gsrc, v4i AS3* lds_dst, imm offset, imm cpol)
    // go through integers: ptrtoint/inttoptr is a clean addrspace conversion;
    // flat LDS address low 32 bits == LDS offset (ISA aperture rule)
    __attribute__((address_space(1))) v4i* gp =
        (__attribute__((address_space(1))) v4i*)(unsigned long long)g;
    __attribute__((address_space(3))) v4i* lp =
        (__attribute__((address_space(3))) v4i*)(unsigned int)(unsigned long long)l;
    __builtin_amdgcn_global_load_async_to_lds_b128(gp, lp, 0, 0);
#else
    *(v8h*)l = *(const v8h*)g;
#endif
}

__device__ __forceinline__ void stage_wait()
{
#if ASYNC_LDS
#if __has_builtin(__builtin_amdgcn_s_wait_asynccnt)
    __builtin_amdgcn_s_wait_asynccnt(0);
#else
    asm volatile("s_wait_asynccnt 0x0" ::: "memory");
#endif
#endif
}

// ---------------------------------------------------------------------------
// WMMA GEMM: C[M,N] = A[M,K] @ Wt[N,K]^T   (fp16 inputs, fp32 accum)
// GATHER: A row m=(b,p,k) built on the fly as [fts_center , fts_nbr - fts_center]
// STATS : write fp16 pre-activations + fused per-channel sum / sumsq atomics
// ---------------------------------------------------------------------------
template<bool GATHER, bool STATS>
__global__ __launch_bounds__(256)
void gemm_wmma_kernel(const _Float16* __restrict__ A,
                      const _Float16* __restrict__ Wt,
                      _Float16* __restrict__ outH,
                      float* __restrict__ outF,
                      float* __restrict__ sum, float* __restrict__ sumsq,
                      int M, int N, int K,
                      const _Float16* __restrict__ fts,
                      const int* __restrict__ knn,
                      int cin, int P, int KNNc)
{
    __shared__ _Float16 As[GEMM_MBLK * LDSW];
    __shared__ _Float16 Bs[GEMM_NBLK * LDSW];
    const int tid  = threadIdx.x;
    const int m0   = blockIdx.x * GEMM_MBLK;
    const int n0   = blockIdx.y * GEMM_NBLK;
    const int w    = tid >> 5;
    const int lane = tid & 31;
    const int wm   = w >> 1;      // 0..3  (32-row strip)
    const int wn   = w & 1;       // 0..1  (32-col strip)
    const int lhalf = lane >> 4;  // 0/1
    const int lmod  = lane & 15;

    v8f acc[2][2] = {};

    // per-thread A-tile load assignment: one row, 16 contiguous halves
    const int arow = tid >> 1;
    const int acg  = (tid & 1) * 16;
    const int gm   = m0 + arow;
    int gbase_p = 0, gbase_q = 0;
    if (GATHER) {
        const int PK  = P * KNNc;
        const int b   = gm / PK;
        const int rem = gm - b * PK;
        const int p   = rem / KNNc;
        const int kk  = rem - p * KNNc;
        gbase_p = b * P + p;
        gbase_q = b * P + knn[gbase_p * KNNc + kk];
    }

    const int ktiles = (K + GEMM_KSTEP - 1) / GEMM_KSTEP;
    for (int kt = 0; kt < ktiles; ++kt) {
        const int k0 = kt * GEMM_KSTEP;
        // ---- stage A tile (128 x 32 halves) ----
        {
            _Float16* dst = &As[arow * LDSW + acg];
            if (GATHER) {
                #pragma unroll 4
                for (int j = 0; j < 16; ++j) {
                    const int c = k0 + acg + j;
                    float v = 0.f;
                    if (c < K) {
                        if (c < cin) {
                            v = (float)fts[(size_t)gbase_p * cin + c];
                        } else {
                            const int cc = c - cin;
                            v = (float)fts[(size_t)gbase_q * cin + cc]
                              - (float)fts[(size_t)gbase_p * cin + cc];
                        }
                    }
                    dst[j] = (_Float16)v;
                }
            } else {
                const _Float16* src = A + (size_t)gm * K + k0 + acg;
                if (k0 + GEMM_KSTEP <= K) {
                    stage16B(src, dst);          // global_load_async_to_lds_b128
                    stage16B(src + 8, dst + 8);
                    if (kt + 1 < ktiles)
                        __builtin_prefetch(src + GEMM_KSTEP, 0, 1); // global_prefetch_b8
                } else {
                    for (int j = 0; j < 16; ++j) {
                        const int c = k0 + acg + j;
                        dst[j] = (c < K) ? src[j] : (_Float16)0.f;
                    }
                }
            }
        }
        // ---- stage B tile (64 rows of Wt x 32 halves) ----
        {
            const int brow = tid >> 2;
            const int bcg  = (tid & 3) * 8;
            _Float16* dst = &Bs[brow * LDSW + bcg];
            const _Float16* src = Wt + (size_t)(n0 + brow) * K + k0 + bcg;
            if (k0 + GEMM_KSTEP <= K) {
                stage16B(src, dst);              // global_load_async_to_lds_b128
            } else {
                for (int j = 0; j < 8; ++j) {
                    const int c = k0 + bcg + j;
                    dst[j] = (c < K) ? src[j] : (_Float16)0.f;
                }
            }
        }
        stage_wait();        // s_wait_asynccnt 0 (per-wave) before the barrier
        __syncthreads();

        // ---- fragments (ISA 7.12.2 16-bit layouts) + 4x WMMA ----
        v16h afrag[2], bfrag[2];
        #pragma unroll
        for (int t = 0; t < 2; ++t) {
            const _Float16* ap = &As[(wm * 32 + t * 16 + lmod) * LDSW];
            v8h lo = *(const v8h*)(ap + lhalf * 8);        // K = kb..kb+7
            v8h hi = *(const v8h*)(ap + 16 + lhalf * 8);   // K = 16+kb..16+kb+7
            afrag[t] = __builtin_shufflevector(lo, hi, 0,1,2,3,4,5,6,7,8,9,10,11,12,13,14,15);
        }
        #pragma unroll
        for (int u = 0; u < 2; ++u) {
            const _Float16* bp = &Bs[(wn * 32 + u * 16 + lmod) * LDSW + lhalf * 8];
            v8h lo = *(const v8h*)bp;                      // K = kb..kb+7
            v8h hi = *(const v8h*)(bp + 16);               // K = 16+kb..16+kb+7
            bfrag[u] = __builtin_shufflevector(lo, hi, 0,1,2,3,4,5,6,7,8,9,10,11,12,13,14,15);
        }
        #pragma unroll
        for (int t = 0; t < 2; ++t)
            #pragma unroll
            for (int u = 0; u < 2; ++u)
                acc[t][u] = __builtin_amdgcn_wmma_f32_16x16x32_f16(
                    false, afrag[t], false, bfrag[u], (short)0, acc[t][u], false, false);
        __syncthreads();
    }

    // ---- epilogue ----
    if (STATS) {
        float s[2] = {0.f, 0.f}, q[2] = {0.f, 0.f};
        #pragma unroll
        for (int t = 0; t < 2; ++t)
            #pragma unroll
            for (int u = 0; u < 2; ++u)
                #pragma unroll
                for (int r = 0; r < 8; ++r) {
                    const int m = m0 + wm * 32 + t * 16 + lhalf * 8 + r;
                    const int n = n0 + wn * 32 + u * 16 + lmod;
                    const float v = acc[t][u][r];
                    outH[(size_t)m * N + n] = (_Float16)v;
                    s[u] += v; q[u] += v * v;
                }
        #pragma unroll
        for (int u = 0; u < 2; ++u) {
            s[u] += __shfl_xor(s[u], 16);
            q[u] += __shfl_xor(q[u], 16);
        }
        const int un = lhalf;                         // lanes 0-15 -> cols 0-15, 16-31 -> 16-31
        const int n  = n0 + wn * 32 + un * 16 + lmod;
        atomicAdd(&sum[n],   s[un]);
        atomicAdd(&sumsq[n], q[un]);
    } else {
        #pragma unroll
        for (int t = 0; t < 2; ++t)
            #pragma unroll
            for (int u = 0; u < 2; ++u)
                #pragma unroll
                for (int r = 0; r < 8; ++r) {
                    const int m = m0 + wm * 32 + t * 16 + lhalf * 8 + r;
                    const int n = n0 + wn * 32 + u * 16 + lmod;
                    outF[(size_t)m * N + n] = acc[t][u][r];
                }
    }
}

// ---------------------------------------------------------------------------
// helpers
// ---------------------------------------------------------------------------
__global__ void zero_f32_kernel(float* p, int n) {
    int i = blockIdx.x * blockDim.x + threadIdx.x;
    if (i < n) p[i] = 0.f;
}

__global__ void w2h_kernel(const float* __restrict__ src, _Float16* __restrict__ dst, int n) {
    int i = blockIdx.x * blockDim.x + threadIdx.x;
    if (i < n) dst[i] = (_Float16)src[i];
}

// mask[b,p], counts[b], fts0[b*P+p][F] = features[b][f][p]*mask
__global__ __launch_bounds__(128)
void mask_fts0_kernel(const float* __restrict__ features, float* __restrict__ mask,
                      float* __restrict__ counts, float* __restrict__ fts0,
                      int B, int F, int P)
{
    const int b = blockIdx.x;
    const int p = threadIdx.x;
    float s = 0.f;
    for (int f = 0; f < F; ++f) s += fabsf(features[(size_t)(b * F + f) * P + p]);
    const float mk = (s != 0.f) ? 1.f : 0.f;
    mask[b * P + p] = mk;
    for (int f = 0; f < F; ++f)
        fts0[((size_t)(b * P + p)) * F + f] = features[(size_t)(b * F + f) * P + p] * mk;
    __shared__ float cs[128];
    cs[p] = mk;
    __syncthreads();
    for (int s2 = 64; s2 >= 1; s2 >>= 1) {
        if (p < s2) cs[p] += cs[p + s2];
        __syncthreads();
    }
    if (p == 0) counts[b] = fmaxf(cs[0], 1.f);
}

__global__ void coords0_kernel(const float* __restrict__ points, const float* __restrict__ mask,
                               _Float16* __restrict__ out, int B, int D, int P)
{
    int i = blockIdx.x * blockDim.x + threadIdx.x;
    if (i >= B * P) return;
    const int b = i / P, p = i - b * P;
    const float mk = mask[i];
    for (int d = 0; d < D; ++d)
        out[(size_t)i * D + d] = (_Float16)(points[(size_t)(b * D + d) * P + p] * mk);
}

// per-channel sum/sumsq of fp32 matrix [M][N]
__global__ __launch_bounds__(256)
void colstats_f32_kernel(const float* __restrict__ X, int M, int N,
                         float* __restrict__ sum, float* __restrict__ sumsq, int rowsPerBlock)
{
    const int m0 = blockIdx.x * rowsPerBlock;
    const int t = threadIdx.x;
    float s0 = 0.f, q0 = 0.f, s1 = 0.f, q1 = 0.f;
    const int mend = min(m0 + rowsPerBlock, M);
    for (int m = m0; m < mend; ++m) {
        const float* row = X + (size_t)m * N;
        if (t < N)       { const float v = row[t];       s0 += v; q0 += v * v; }
        if (t + 256 < N) { const float v = row[t + 256]; s1 += v; q1 += v * v; }
    }
    if (t < N)       { atomicAdd(&sum[t], s0);       atomicAdd(&sumsq[t], q0); }
    if (t + 256 < N) { atomicAdd(&sum[t + 256], s1); atomicAdd(&sumsq[t + 256], q1); }
}

__global__ void bn_finalize_kernel(const float* __restrict__ sum, const float* __restrict__ sumsq,
                                   const float* __restrict__ g, const float* __restrict__ bb,
                                   float invM, int N, float* __restrict__ scale, float* __restrict__ bias)
{
    int n = blockIdx.x * blockDim.x + threadIdx.x;
    if (n >= N) return;
    const float mean = sum[n] * invM;
    const float var  = sumsq[n] * invM - mean * mean;
    const float sc = g[n] * rsqrtf(var + 1e-5f);
    scale[n] = sc;
    bias[n]  = bb[n] - mean * sc;
}

__global__ void bn_mask_f32_to_f16_kernel(const float* __restrict__ X, const float* __restrict__ scale,
                                          const float* __restrict__ bias, const float* __restrict__ mask,
                                          _Float16* __restrict__ out, int M, int N)
{
    size_t i = (size_t)blockIdx.x * blockDim.x + threadIdx.x;
    if (i >= (size_t)M * N) return;
    const int m = (int)(i / N), n = (int)(i - (size_t)m * N);
    out[i] = (_Float16)((X[i] * scale[n] + bias[n]) * mask[m]);
}

__global__ void bn_relu_f16_inplace_kernel(_Float16* __restrict__ X, const float* __restrict__ scale,
                                           const float* __restrict__ bias, size_t MN, int N)
{
    size_t i = (size_t)blockIdx.x * blockDim.x + threadIdx.x;
    if (i >= MN) return;
    const int n = (int)(i % (size_t)N);
    const float v = (float)X[i] * scale[n] + bias[n];
    X[i] = (_Float16)fmaxf(v, 0.f);
}

// one block per (b,p): d2 row in LDS, 17 argmin rounds (top_k tie: lowest index)
__global__ __launch_bounds__(128)
void knn_kernel(const _Float16* __restrict__ coords, const float* __restrict__ mask,
                int* __restrict__ idxout, int P, int C)
{
    __shared__ float cp[128];
    __shared__ float d2[128];
    __shared__ float rv[64];
    __shared__ int   ri[64];
    const int b = blockIdx.x / P;
    const int p = blockIdx.x - b * P;
    const int t = threadIdx.x;  // = q
    for (int c = t; c < C; c += 128)
        cp[c] = (float)coords[((size_t)(b * P + p)) * C + c];
    __syncthreads();
    const float sp = (mask[b * P + p] == 0.f) ? 1e9f : 0.f;
    const float sq = (mask[b * P + t] == 0.f) ? 1e9f : 0.f;
    const float ds = sp - sq;
    float acc = 0.f;
    const _Float16* cq = coords + ((size_t)(b * P + t)) * C;
    for (int c = 0; c < C; ++c) {
        const float diff = cp[c] - (float)cq[c] + ds;
        acc += diff * diff;
    }
    d2[t] = acc;
    __syncthreads();
    for (int r = 0; r < 17; ++r) {
        if (t < 64) {
            float v1 = d2[t], v2 = d2[t + 64];
            if (v2 < v1) { rv[t] = v2; ri[t] = t + 64; } else { rv[t] = v1; ri[t] = t; }
        }
        __syncthreads();
        for (int s = 32; s >= 1; s >>= 1) {
            if (t < s) {
                const float v2 = rv[t + s]; const int i2 = ri[t + s];
                if (v2 < rv[t] || (v2 == rv[t] && i2 < ri[t])) { rv[t] = v2; ri[t] = i2; }
            }
            __syncthreads();
        }
        if (t == 0) {
            const int bi = ri[0];
            if (r > 0) idxout[((size_t)(b * P + p)) * 16 + (r - 1)] = bi;
            d2[bi] = 3.0e38f;
        }
        __syncthreads();
    }
}

// mean over K of relu(bn(conv3 pre-act))
__global__ void agg_kernel(const _Float16* __restrict__ X, const float* __restrict__ scale,
                           const float* __restrict__ bias, float* __restrict__ agg,
                           int KNNc, int N)
{
    const int row = blockIdx.x;  // b*P+p
    const int n = threadIdx.x;
    if (n >= N) return;
    float s = 0.f;
    for (int k = 0; k < KNNc; ++k) {
        const float v = (float)X[((size_t)row * KNNc + k) * N + n] * scale[n] + bias[n];
        s += fmaxf(v, 0.f);
    }
    agg[(size_t)row * N + n] = s / (float)KNNc;
}

// fts_new = relu(bn(sc) + agg) * mask  ->  fts buffer + concat slice (fp16)
__global__ void combine_kernel(const float* __restrict__ SC, const float* __restrict__ scale,
                               const float* __restrict__ bias, const float* __restrict__ agg,
                               const float* __restrict__ mask, _Float16* __restrict__ ftsOut,
                               _Float16* __restrict__ concatOut, int N, int cStride, int cOff)
{
    const int row = blockIdx.x;
    const int n = threadIdx.x;
    if (n >= N) return;
    const float v = fmaxf(SC[(size_t)row * N + n] * scale[n] + bias[n] + agg[(size_t)row * N + n], 0.f)
                  * mask[row];
    ftsOut[(size_t)row * N + n] = (_Float16)v;
    concatOut[(size_t)row * cStride + cOff + n] = (_Float16)v;
}

// relu(bn(x))*mask, sum over P, / counts
__global__ void fusion_pool_kernel(const float* __restrict__ X, const float* __restrict__ scale,
                                   const float* __restrict__ bias, const float* __restrict__ mask,
                                   const float* __restrict__ counts, float* __restrict__ pooled,
                                   int P, int N)
{
    const int b = blockIdx.x;
    const int n = threadIdx.x;
    if (n >= N) return;
    float s = 0.f;
    for (int p = 0; p < P; ++p) {
        const int row = b * P + p;
        s += fmaxf(X[(size_t)row * N + n] * scale[n] + bias[n], 0.f) * mask[row];
    }
    pooled[(size_t)b * N + n] = s / counts[b];
}

__global__ void fc_kernel(const float* __restrict__ X, const float* __restrict__ W,
                          const float* __restrict__ bias, float* __restrict__ out,
                          int Bn, int K, int N, int relu)
{
    int i = blockIdx.x * blockDim.x + threadIdx.x;
    if (i >= Bn * N) return;
    const int b = i / N, n = i - b * N;
    float s = bias[n];
    for (int k = 0; k < K; ++k) s += X[(size_t)b * K + k] * W[(size_t)n * K + k];
    out[i] = relu ? fmaxf(s, 0.f) : s;
}

// ---------------------------------------------------------------------------
// orchestration
// ---------------------------------------------------------------------------
// param leaf indices assuming JAX pytree (sorted-dict-key) flatten order:
// blocks(0..2): [bn0_b,bn0_g,bn1_b,bn1_g,bn2_b,bn2_g,conv0,conv1,conv2,scbn_b,scbn_g,sc_w]
static inline int blk_base(int b) { return 2 + b * 12; }

extern "C" void kernel_launch(void* const* d_in, const int* in_sizes, int n_in,
                              void* d_out, int out_size, void* d_ws, size_t ws_size,
                              hipStream_t stream)
{
    (void)in_sizes; (void)n_in; (void)out_size; (void)ws_size;
    const int B = 128, P = 128, D = 2, F = 16, KNN = 16;
    const int BP = B * P;            // 16384
    const int ME = BP * KNN;         // 262144
    const int cins[3]  = {16, 64, 128};
    const int couts[3] = {64, 128, 256};
    const int CONCAT = 448, FUSN = 384;

    const float* points   = (const float*)d_in[0];
    const float* features = (const float*)d_in[1];
    auto PF = [&](int i) { return (const float*)d_in[i]; };
    const int IDX_BNFTS_B = 38, IDX_BNFTS_G = 39;
    const int IDX_FC1_B = 40, IDX_FC1_W = 41;
    const int IDX_FCO_B = 42, IDX_FCO_W = 43;
    const int IDX_FUS_BN_B = 44, IDX_FUS_BN_G = 45, IDX_FUS_W = 46;

    // ---- bump allocator on d_ws ----
    char* base = (char*)d_ws;
    size_t off = 0;
    auto alloc = [&](size_t bytes) -> void* {
        void* p = base + off;
        off += (bytes + 255) & ~(size_t)255;
        return p;
    };
    _Float16* fts_a   = (_Float16*)alloc((size_t)BP * 256 * 2);
    _Float16* fts_b   = (_Float16*)alloc((size_t)BP * 256 * 2);
    _Float16* concatb = (_Float16*)alloc((size_t)BP * CONCAT * 2);
    float*    maskb   = (float*)alloc((size_t)BP * 4);
    float*    countsb = (float*)alloc((size_t)B * 4);
    float*    fts0b   = (float*)alloc((size_t)BP * F * 4);
    _Float16* coords0 = (_Float16*)alloc((size_t)BP * D * 2);
    int*      idxb    = (int*)alloc((size_t)BP * KNN * 4);
    _Float16* X1      = (_Float16*)alloc((size_t)ME * 256 * 2);
    _Float16* X2      = (_Float16*)alloc((size_t)ME * 256 * 2);
    float*    SCF     = (float*)alloc((size_t)BP * FUSN * 4);
    float*    aggb    = (float*)alloc((size_t)BP * 256 * 4);
    float*    pooled  = (float*)alloc((size_t)B * FUSN * 4);
    float*    hbuf    = (float*)alloc((size_t)B * 256 * 4);
    float*    statb   = (float*)alloc(2048 * 4);
    float* sum = statb, *sumsq = statb + 512, *scl = statb + 1024, *bia = statb + 1536;

    // ---- convert weights fp32 -> fp16 (Wt layout = W[n][k], already row-major) ----
    _Float16* Wh[3][3]; _Float16* WhSC[3]; _Float16* WhFus;
    for (int b = 0; b < 3; ++b) {
        for (int j = 0; j < 3; ++j) {
            const int n = couts[b] * ((j == 0) ? 2 * cins[b] : couts[b]);
            Wh[b][j] = (_Float16*)alloc((size_t)n * 2);
            w2h_kernel<<<(n + 255) / 256, 256, 0, stream>>>(PF(blk_base(b) + 6 + j), Wh[b][j], n);
        }
        const int n = couts[b] * cins[b];
        WhSC[b] = (_Float16*)alloc((size_t)n * 2);
        w2h_kernel<<<(n + 255) / 256, 256, 0, stream>>>(PF(blk_base(b) + 11), WhSC[b], n);
    }
    {
        const int n = FUSN * CONCAT;
        WhFus = (_Float16*)alloc((size_t)n * 2);
        w2h_kernel<<<(n + 255) / 256, 256, 0, stream>>>(PF(IDX_FUS_W), WhFus, n);
    }

    auto zero_stats = [&]() {
        zero_f32_kernel<<<4, 256, 0, stream>>>(statb, 1024);
    };
    auto finalize = [&](const float* g, const float* bb, float invM, int N) {
        bn_finalize_kernel<<<(N + 63) / 64, 64, 0, stream>>>(sum, sumsq, g, bb, invM, N, scl, bia);
    };

    // ---- stage 0: mask, counts, initial BN of features ----
    mask_fts0_kernel<<<B, P, 0, stream>>>(features, maskb, countsb, fts0b, B, F, P);
    coords0_kernel<<<(BP + 255) / 256, 256, 0, stream>>>(points, maskb, coords0, B, D, P);
    zero_stats();
    colstats_f32_kernel<<<(BP + 63) / 64, 256, 0, stream>>>(fts0b, BP, F, sum, sumsq, 64);
    finalize(PF(IDX_BNFTS_G), PF(IDX_BNFTS_B), 1.f / BP, F);
    bn_mask_f32_to_f16_kernel<<<(BP * F + 255) / 256, 256, 0, stream>>>(fts0b, scl, bia, maskb, fts_a, BP, F);

    // ---- EdgeConv blocks ----
    _Float16* fcur = fts_a;
    _Float16* fnext = fts_b;
    int cOff = 0;
    for (int blk = 0; blk < 3; ++blk) {
        const int ci = cins[blk], co = couts[blk];
        const _Float16* coords = (blk == 0) ? coords0 : fcur;
        const int cdim = (blk == 0) ? D : ci;
        knn_kernel<<<BP, 128, 0, stream>>>(coords, maskb, idxb, P, cdim);

        dim3 gE(ME / GEMM_MBLK, co / GEMM_NBLK);
        // conv0 (edge-gather A)
        zero_stats();
        gemm_wmma_kernel<true, true><<<gE, 256, 0, stream>>>(
            nullptr, Wh[blk][0], X1, nullptr, sum, sumsq, ME, co, 2 * ci, fcur, idxb, ci, P, KNN);
        finalize(PF(blk_base(blk) + 1), PF(blk_base(blk) + 0), 1.f / ME, co);
        bn_relu_f16_inplace_kernel<<<(int)(((size_t)ME * co + 255) / 256), 256, 0, stream>>>(
            X1, scl, bia, (size_t)ME * co, co);
        // conv1
        zero_stats();
        gemm_wmma_kernel<false, true><<<gE, 256, 0, stream>>>(
            X1, Wh[blk][1], X2, nullptr, sum, sumsq, ME, co, co, nullptr, nullptr, 0, P, KNN);
        finalize(PF(blk_base(blk) + 3), PF(blk_base(blk) + 2), 1.f / ME, co);
        bn_relu_f16_inplace_kernel<<<(int)(((size_t)ME * co + 255) / 256), 256, 0, stream>>>(
            X2, scl, bia, (size_t)ME * co, co);
        // conv2
        zero_stats();
        gemm_wmma_kernel<false, true><<<gE, 256, 0, stream>>>(
            X2, Wh[blk][2], X1, nullptr, sum, sumsq, ME, co, co, nullptr, nullptr, 0, P, KNN);
        finalize(PF(blk_base(blk) + 5), PF(blk_base(blk) + 4), 1.f / ME, co);
        // aggregate mean over K (applies bn+relu of conv2)
        agg_kernel<<<BP, co, 0, stream>>>(X1, scl, bia, aggb, KNN, co);
        // shortcut GEMM (fp32 out) + its BN stats
        dim3 gS(BP / GEMM_MBLK, co / GEMM_NBLK);
        gemm_wmma_kernel<false, false><<<gS, 256, 0, stream>>>(
            fcur, WhSC[blk], nullptr, SCF, nullptr, nullptr, BP, co, ci, nullptr, nullptr, 0, P, KNN);
        zero_stats();
        colstats_f32_kernel<<<(BP + 63) / 64, 256, 0, stream>>>(SCF, BP, co, sum, sumsq, 64);
        finalize(PF(blk_base(blk) + 10), PF(blk_base(blk) + 9), 1.f / BP, co);
        // combine -> new fts + concat slice
        combine_kernel<<<BP, co, 0, stream>>>(SCF, scl, bia, aggb, maskb, fnext, concatb, co, CONCAT, cOff);
        cOff += co;
        _Float16* tmp = fcur; fcur = fnext; fnext = tmp;
    }

    // ---- fusion + pooling + FC ----
    dim3 gF(BP / GEMM_MBLK, FUSN / GEMM_NBLK);
    gemm_wmma_kernel<false, false><<<gF, 256, 0, stream>>>(
        concatb, WhFus, nullptr, SCF, nullptr, nullptr, BP, FUSN, CONCAT, nullptr, nullptr, 0, P, KNN);
    zero_stats();
    colstats_f32_kernel<<<(BP + 63) / 64, 256, 0, stream>>>(SCF, BP, FUSN, sum, sumsq, 64);
    finalize(PF(IDX_FUS_BN_G), PF(IDX_FUS_BN_B), 1.f / BP, FUSN);
    fusion_pool_kernel<<<B, FUSN, 0, stream>>>(SCF, scl, bia, maskb, countsb, pooled, P, FUSN);
    fc_kernel<<<(B * 256 + 255) / 256, 256, 0, stream>>>(pooled, PF(IDX_FC1_W), PF(IDX_FC1_B), hbuf, B, FUSN, 256, 1);
    fc_kernel<<<(B * 5 + 255) / 256, 256, 0, stream>>>(hbuf, PF(IDX_FCO_W), PF(IDX_FCO_B), (float*)d_out, B, 256, 5, 0);
}
